// SOLOv2_33732673143018
// MI455X (gfx1250) — 8-bit matrix operand order repaired and checked by reading, compiled, not run on hardware
//
#include <hip/hip_runtime.h>
#include <math.h>

#define N_MASKS   500
#define NT        512
#define HW        65536
#define NMS_SIGMA 2.0f

typedef __attribute__((ext_vector_type(8))) int v8i;

#define WS_PACKED    0
#define WS_SCORES_U  33554432
#define WS_LABELS_U  33556480
#define WS_SUMS_U    33558528
#define WS_ORDER     33560576
#define WS_SCORES_S  33562624
#define WS_LABELS_S  33564672
#define WS_SUMS_S    33566720
#define WS_CMAX      33568768
#define WS_INTER     33570816
#define WS_DECAY     34619392

__global__ __launch_bounds__(256)
void k_prep(const float* __restrict__ mask,
            const float* __restrict__ cate_scores,
            const int*   __restrict__ cate_labels,
            unsigned char* __restrict__ packed,
            float* __restrict__ scoresU,
            int*   __restrict__ labelsU,
            float* __restrict__ sumsU) {
  const int n = blockIdx.x;
  const int t = threadIdx.x;
  unsigned int* row = (unsigned int*)(packed + (size_t)n * HW);

  if (n < N_MASKS) {
    const float4* mrow = (const float4*)(mask + (size_t)n * HW);
    float soft = 0.0f;
    float cnt  = 0.0f;
    #pragma unroll 4
    for (int it = 0; it < 64; ++it) {
      const int idx = t + it * 256;
      const float4 p = mrow[idx];
      const unsigned int b0 = p.x > 0.5f ? 1u : 0u;
      const unsigned int b1 = p.y > 0.5f ? 1u : 0u;
      const unsigned int b2 = p.z > 0.5f ? 1u : 0u;
      const unsigned int b3 = p.w > 0.5f ? 1u : 0u;
      row[idx] = b0 | (b1 << 8) | (b2 << 16) | (b3 << 24);
      soft += (b0 ? p.x : 0.0f) + (b1 ? p.y : 0.0f) +
              (b2 ? p.z : 0.0f) + (b3 ? p.w : 0.0f);
      cnt  += (float)(b0 + b1 + b2 + b3);
    }
    __shared__ float s_soft[256];
    __shared__ float s_cnt[256];
    s_soft[t] = soft; s_cnt[t] = cnt;
    __syncthreads();
    for (int s = 128; s > 0; s >>= 1) {
      if (t < s) { s_soft[t] += s_soft[t + s]; s_cnt[t] += s_cnt[t + s]; }
      __syncthreads();
    }
    if (t == 0) {
      const float c   = s_cnt[0];
      const float seg = s_soft[0] / fmaxf(c, 1.0f);
      scoresU[n] = cate_scores[n] * seg;
      labelsU[n] = cate_labels[n];
      sumsU[n]   = c;
    }
  } else {
    uint4 z = make_uint4(0u, 0u, 0u, 0u);
    uint4* row4 = (uint4*)row;
    #pragma unroll
    for (int it = 0; it < 16; ++it) row4[t + it * 256] = z;
    if (t == 0) {
      scoresU[n] = -INFINITY;
      labelsU[n] = -1;
      sumsU[n]   = 0.0f;
    }
  }
}

__global__ __launch_bounds__(512)
void k_sort(const float* __restrict__ scoresU,
            const int*   __restrict__ labelsU,
            const float* __restrict__ sumsU,
            int*   __restrict__ order,
            float* __restrict__ scoresS,
            int*   __restrict__ labelsS,
            float* __restrict__ sumsS) {
  __shared__ float key[NT];
  __shared__ int   idx[NT];
  const int t = threadIdx.x;
  key[t] = -scoresU[t];
  idx[t] = t;
  __syncthreads();

  for (int k = 2; k <= NT; k <<= 1) {
    for (int j = k >> 1; j > 0; j >>= 1) {
      const int ixj = t ^ j;
      if (ixj > t) {
        const bool asc = ((t & k) == 0);
        const float ka = key[t], kb = key[ixj];
        const int   ia = idx[t], ib = idx[ixj];
        const bool aGreater = (ka > kb) || (ka == kb && ia > ib);
        if (asc ? aGreater : !aGreater) {
          key[t] = kb; key[ixj] = ka;
          idx[t] = ib; idx[ixj] = ia;
        }
      }
      __syncthreads();
    }
  }
  const int src = idx[t];
  order[t]   = src;
  scoresS[t] = scoresU[src];
  labelsS[t] = labelsU[src];
  sumsS[t]   = sumsU[src];
}

__global__ __launch_bounds__(256)
void k_gemm(const unsigned char* __restrict__ packed,
            const int* __restrict__ order,
            int* __restrict__ inter) {
  const int t      = threadIdx.x;
  const int lane   = t & 31;
  const int wid    = t >> 5;
  const int w      = blockIdx.x * 8 + wid;
  const int ti     = w >> 3;
  const int tj     = w & 7;
  const int lane16 = lane & 15;
  const int hi     = lane >> 4;

  const int i0 = ti * 32;
  const int j0 = tj * 64;

  const unsigned char* aBase0 =
      packed + (size_t)order[i0 + lane16] * HW + (size_t)(hi * 16);
  const unsigned char* aBase1 =
      packed + (size_t)order[i0 + 16 + lane16] * HW + (size_t)(hi * 16);
  const unsigned char* bBase[4];
  #pragma unroll
  for (int jt = 0; jt < 4; ++jt)
    bBase[jt] = packed + (size_t)order[j0 + jt * 16 + lane16] * HW +
                (size_t)(hi * 16);

  v8i acc[2][4] = {};

  for (int k = 0; k < HW; k += 64) {
    const uint2 a00 = *(const uint2*)(aBase0 + k);
    const uint2 a01 = *(const uint2*)(aBase0 + k + 8);
    const uint2 a02 = *(const uint2*)(aBase0 + k + 32);
    const uint2 a03 = *(const uint2*)(aBase0 + k + 40);
    const v8i a0 = {(int)a00.x, (int)a00.y, (int)a01.x, (int)a01.y,
                    (int)a02.x, (int)a02.y, (int)a03.x, (int)a03.y};

    const uint2 a10 = *(const uint2*)(aBase1 + k);
    const uint2 a11 = *(const uint2*)(aBase1 + k + 8);
    const uint2 a12 = *(const uint2*)(aBase1 + k + 32);
    const uint2 a13 = *(const uint2*)(aBase1 + k + 40);
    const v8i a1 = {(int)a10.x, (int)a10.y, (int)a11.x, (int)a11.y,
                    (int)a12.x, (int)a12.y, (int)a13.x, (int)a13.y};

    #pragma unroll
    for (int jt = 0; jt < 4; ++jt) {
      const uint4 bl = *(const uint4*)(bBase[jt] + k);
      const uint4 bh = *(const uint4*)(bBase[jt] + k + 32);
      const v8i b = {(int)bl.x, (int)bl.y, (int)bl.z, (int)bl.w,
                     (int)bh.x, (int)bh.y, (int)bh.z, (int)bh.w};
      acc[0][jt] = __builtin_amdgcn_wmma_i32_16x16x64_iu8(
          false, a0, false, b, acc[0][jt], false, false);
      acc[1][jt] = __builtin_amdgcn_wmma_i32_16x16x64_iu8(
          false, a1, false, b, acc[1][jt], false, false);
    }
  }

  #pragma unroll
  for (int mt = 0; mt < 2; ++mt)
    #pragma unroll
    for (int jt = 0; jt < 4; ++jt)
      #pragma unroll
      for (int r = 0; r < 8; ++r) {
        const int row = i0 + mt * 16 + r + hi * 8;
        const int col = j0 + jt * 16 + lane16;
        inter[row * NT + col] = acc[mt][jt][r];
      }
}

__global__ __launch_bounds__(256)
void k_decay(const int*   __restrict__ inter,
             const int*   __restrict__ labelsS,
             const float* __restrict__ sumsS,
             float* __restrict__ decayM,
             float* __restrict__ cmax) {
  const int j = blockIdx.x;
  const int t = threadIdx.x;
  const int   lj = labelsS[j];
  const float sj = sumsS[j];
  float lmax = 0.0f;
  for (int i = t; i < NT; i += 256) {
    float d = 0.0f;
    if (i < j && j < N_MASKS && i < N_MASKS && labelsS[i] == lj) {
      const float iv = (float)inter[i * NT + j];
      const float un = sumsS[i] + sj - iv;
      d = iv / fmaxf(un, 1.0f);
    }
    decayM[i * NT + j] = d;
    lmax = fmaxf(lmax, d);
  }
  __shared__ float sm[256];
  sm[t] = lmax;
  __syncthreads();
  for (int s = 128; s > 0; s >>= 1) {
    if (t < s) sm[t] = fmaxf(sm[t], sm[t + s]);
    __syncthreads();
  }
  if (t == 0) cmax[j] = sm[0];
}

__global__ __launch_bounds__(256)
void k_final(const float* __restrict__ decayM,
             const float* __restrict__ cmax,
             const float* __restrict__ scoresS,
             float* __restrict__ out) {
  const int j = blockIdx.x;
  const int t = threadIdx.x;
  float lmax = -INFINITY;
  for (int i = t; i < N_MASKS; i += 256) {
    const float d = decayM[i * NT + j];
    const float c = cmax[i];
    lmax = fmaxf(lmax, d * d - c * c);
  }
  __shared__ float sm[256];
  sm[t] = lmax;
  __syncthreads();
  for (int s = 128; s > 0; s >>= 1) {
    if (t < s) sm[t] = fmaxf(sm[t], sm[t + s]);
    __syncthreads();
  }
  if (t == 0) out[j] = scoresS[j] * expf(-NMS_SIGMA * sm[0]);
}

extern "C" void kernel_launch(void* const* d_in, const int* in_sizes, int n_in,
                              void* d_out, int out_size, void* d_ws, size_t ws_size,
                              hipStream_t stream) {
  const float* mask_preds  = (const float*)d_in[0];
  const float* cate_scores = (const float*)d_in[1];
  const int*   cate_labels = (const int*)d_in[2];
  float* out = (float*)d_out;

  char* ws = (char*)d_ws;
  unsigned char* packed  = (unsigned char*)(ws + WS_PACKED);
  float* scoresU = (float*)(ws + WS_SCORES_U);
  int*   labelsU = (int*)  (ws + WS_LABELS_U);
  float* sumsU   = (float*)(ws + WS_SUMS_U);
  int*   order   = (int*)  (ws + WS_ORDER);
  float* scoresS = (float*)(ws + WS_SCORES_S);
  int*   labelsS = (int*)  (ws + WS_LABELS_S);
  float* sumsS   = (float*)(ws + WS_SUMS_S);
  float* cmax    = (float*)(ws + WS_CMAX);
  int*   inter   = (int*)  (ws + WS_INTER);
  float* decayM  = (float*)(ws + WS_DECAY);

  k_prep <<<NT,       256, 0, stream>>>(mask_preds, cate_scores, cate_labels,
                                        packed, scoresU, labelsU, sumsU);
  k_sort <<<1,        512, 0, stream>>>(scoresU, labelsU, sumsU,
                                        order, scoresS, labelsS, sumsS);
  k_gemm <<<16,       256, 0, stream>>>(packed, order, inter);
  k_decay<<<NT,       256, 0, stream>>>(inter, labelsS, sumsS, decayM, cmax);
  k_final<<<N_MASKS,  256, 0, stream>>>(decayM, cmax, scoresS, out);
}
